// ShapeOptNet_72215580115691
// MI455X (gfx1250) — matrix operand order repaired and checked by reading, not compiled, not run on hardware
//
#include <hip/hip_runtime.h>
#include <hip/hip_bf16.h>

typedef __attribute__((ext_vector_type(16))) _Float16 v16h;
typedef __attribute__((ext_vector_type(8)))  _Float16 v8h;
typedef __attribute__((ext_vector_type(8)))  float    v8f;

// ---------------------------------------------------------------------------
// conv1 (3->8, 3x3, stride 2) + ReLU + 2x2 maxpool, fused. f32 math (HBM-bound:
// reads the 99MB input once). Output: act f16 NHWC [8][254][254][32], channels
// 8..31 zero-padded so the WMMA conv2 can consume 32-wide K chunks unmasked.
// ---------------------------------------------------------------------------
__global__ void conv1_pool(const float* __restrict__ seg,
                           const float* __restrict__ w,
                           const float* __restrict__ b,
                           _Float16* __restrict__ act1) {
  __shared__ float sw[216];                   // 8*3*3*3 weights
  int tid = threadIdx.x;
  if (tid < 216) sw[tid] = w[tid];
  __syncthreads();
  int idx = blockIdx.x * blockDim.x + tid;
  const int HP = 254;
  if (idx >= 8 * HP * HP) return;
  int wp = idx % HP; int t = idx / HP; int hp = t % HP; int f = t / HP;
  const float* in = seg + (size_t)f * 3 * 1017 * 1017;
  float acc[4][8];
  for (int q = 0; q < 4; ++q)
    for (int co = 0; co < 8; ++co) acc[q][co] = b[co];
  for (int q = 0; q < 4; ++q) {
    int ho = 2 * hp + (q >> 1);
    int wo = 2 * wp + (q & 1);
    int hi0 = 2 * ho, wi0 = 2 * wo;          // stride 2
    for (int ci = 0; ci < 3; ++ci)
      for (int ky = 0; ky < 3; ++ky)
        for (int kx = 0; kx < 3; ++kx) {
          float v = in[(size_t)ci * 1017 * 1017 + (size_t)(hi0 + ky) * 1017 + (wi0 + kx)];
          for (int co = 0; co < 8; ++co)
            acc[q][co] += v * sw[((co * 3 + ci) * 3 + ky) * 3 + kx];
        }
  }
  _Float16* o = act1 + (size_t)idx * 32;
  for (int c = 0; c < 32; ++c) {
    float m = 0.f;
    if (c < 8) {
      m = fmaxf(fmaxf(acc[0][c], acc[1][c]), fmaxf(acc[2][c], acc[3][c]));
      m = fmaxf(m, 0.f);                     // pool(relu(x)) == relu(pool(x))
    }
    o[c] = (_Float16)m;
  }
}

// ---------------------------------------------------------------------------
// Repack OIHW f32 conv weights -> [tap(9)][cout][cin_pad] f16 (B-matrix rows).
// ---------------------------------------------------------------------------
__global__ void repack_conv_w(const float* __restrict__ w, _Float16* __restrict__ wp,
                              int Cin, int Cinp, int Cout) {
  int idx = blockIdx.x * blockDim.x + threadIdx.x;
  int total = 9 * Cout * Cinp;
  if (idx >= total) return;
  int ci = idx % Cinp; int t = idx / Cinp; int co = t % Cout; int tap = t / Cout;
  int ty = tap / 3, tx = tap % 3;
  float v = (ci < Cin) ? w[((co * Cin + ci) * 3 + ty) * 3 + tx] : 0.f;
  wp[idx] = (_Float16)v;
}

// ---------------------------------------------------------------------------
// Implicit-GEMM 3x3 conv via v_wmma_f32_16x16x32_f16.
//   M tile = 16 output pixels along W, N tile = 16 output channels,
//   K loop  = 9 taps x (Cinp/32) chunks.
// A fragment per ISA 16-bit A layout: lanes 0-15 hold K {0..7,16..23},
// lanes 16-31 hold K {8..15,24..31} for row M = lane&15.
// ---------------------------------------------------------------------------
__launch_bounds__(32)
__global__ void conv_wmma(const _Float16* __restrict__ act,
                          const _Float16* __restrict__ wpk,
                          const float* __restrict__ bias,
                          _Float16* __restrict__ out,
                          int Hin, int Cinp, int Cout, int Hout, int dorelu) {
  const int lane   = threadIdx.x & 31;
  const int lrow   = lane & 15;
  const int khalf  = lane >> 4;
  const int ntiles = Cout >> 4;
  const int frame  = blockIdx.z / ntiles;
  const int nbase  = (blockIdx.z - frame * ntiles) << 4;
  const int ho     = blockIdx.y;
  const int wo     = (blockIdx.x << 4) + lrow;
  const int woc    = (wo < Hout) ? wo : (Hout - 1);      // clamp; masked on store
  const _Float16* A = act + (size_t)frame * Hin * Hin * Cinp;
  const int kch = Cinp >> 5;
  v8f acc = {};
  for (int ty = 0; ty < 3; ++ty) {
    for (int tx = 0; tx < 3; ++tx) {
      const _Float16* arow = A + ((size_t)(ho + ty) * Hin + (size_t)(woc + tx)) * Cinp;
      const _Float16* brow = wpk + ((size_t)((ty * 3 + tx) * Cout) + nbase + lrow) * Cinp;
      __builtin_prefetch(arow + (size_t)Hin * Cinp, 0, 1);  // global_prefetch_b8
      for (int cb = 0; cb < kch; ++cb) {
        v8h a0 = *(const v8h*)(arow + cb * 32 + khalf * 8);
        v8h a1 = *(const v8h*)(arow + cb * 32 + khalf * 8 + 16);
        v16h a = __builtin_shufflevector(a0, a1, 0,1,2,3,4,5,6,7,8,9,10,11,12,13,14,15);
        v8h b0 = *(const v8h*)(brow + cb * 32 + khalf * 8);
        v8h b1 = *(const v8h*)(brow + cb * 32 + khalf * 8 + 16);
        v16h bm = __builtin_shufflevector(b0, b1, 0,1,2,3,4,5,6,7,8,9,10,11,12,13,14,15);
        acc = __builtin_amdgcn_wmma_f32_16x16x32_f16(false, a, false, bm,
                                                     (short)0, acc, false, false);
      }
    }
  }
  const float bn = bias[nbase + lrow];                    // this lane's N column
  _Float16* O = out + (size_t)frame * Hout * Hout * Cout;
  for (int r = 0; r < 8; ++r) {
    int m  = r + khalf * 8;                               // D row per ISA layout
    int w2 = (blockIdx.x << 4) + m;
    if (w2 < Hout) {
      float v = acc[r] + bn;
      if (dorelu) v = fmaxf(v, 0.f);
      O[((size_t)ho * Hout + w2) * Cout + nbase + lrow] = (_Float16)v;
    }
  }
}

// ---------------------------------------------------------------------------
// 2x2 maxpool + zero channel-pad. raw [Hc][Hc][Cout] f16 -> act [Hp][Hp][Cpad].
// ---------------------------------------------------------------------------
__global__ void pool2(const _Float16* __restrict__ raw, _Float16* __restrict__ act,
                      int Hc, int Cout, int Hp, int Cpad) {
  size_t idx = blockIdx.x * (size_t)blockDim.x + threadIdx.x;
  size_t total = (size_t)8 * Hp * Hp * Cpad;
  if (idx >= total) return;
  int c = idx % Cpad; size_t t = idx / Cpad;
  int wpp = t % Hp; t /= Hp; int hp = t % Hp; int f = (int)(t / Hp);
  _Float16 v = (_Float16)0.f;
  if (c < Cout) {
    const _Float16* R = raw + (size_t)f * Hc * Hc * Cout;
    float m = -1e30f;
    for (int dy = 0; dy < 2; ++dy)
      for (int dx = 0; dx < 2; ++dx)
        m = fmaxf(m, (float)R[((size_t)(2 * hp + dy) * Hc + (2 * wpp + dx)) * Cout + c]);
    v = (_Float16)m;
  }
  act[idx] = v;
}

// flat = transpose(c,(0,3,2,1)).reshape : flat[w*14*128 + h*128 + c] = act5[h][w][c]
__global__ void flatten5(const _Float16* __restrict__ act5, _Float16* __restrict__ flatA) {
  int idx = blockIdx.x * blockDim.x + threadIdx.x;
  if (idx >= 8 * 25088) return;
  int i = idx % 25088; int f = idx / 25088;
  int c = i % 128; int hw = i / 128; int h = hw % 14; int w = hw / 14;
  flatA[idx] = act5[(((size_t)f * 14 + h) * 14 + w) * 128 + c];
}

// fc1 (20x25088) and fc4 (200x25088) concatenated into one f16 B matrix, 224 rows.
__global__ void repack_wcat(const float* __restrict__ fc1w, const float* __restrict__ fc4w,
                            _Float16* __restrict__ wcat) {
  size_t idx = blockIdx.x * (size_t)blockDim.x + threadIdx.x;
  if (idx >= (size_t)224 * 25088) return;
  int k = (int)(idx % 25088); int n = (int)(idx / 25088);
  float v = 0.f;
  if (n < 20)       v = fc1w[(size_t)n * 25088 + k];
  else if (n < 220) v = fc4w[(size_t)(n - 20) * 25088 + k];
  wcat[idx] = (_Float16)v;
}

// ---------------------------------------------------------------------------
// fc1+fc4 as one WMMA GEMM: M=16 (8 frames + zero rows), N=224, K=25088.
// ---------------------------------------------------------------------------
__launch_bounds__(32)
__global__ void fcgemm(const _Float16* __restrict__ flatA,
                       const _Float16* __restrict__ wcat,
                       const float* __restrict__ fc1b,
                       const float* __restrict__ fc4b,
                       float* __restrict__ shape20,
                       float* __restrict__ pose200) {
  const int lane  = threadIdx.x & 31;
  const int lrow  = lane & 15;
  const int khalf = lane >> 4;
  const int nbase = blockIdx.x << 4;
  const int n     = nbase + lrow;
  const _Float16* arow = flatA + (size_t)lrow * 25088;   // A row = frame
  const _Float16* brow = wcat + (size_t)n * 25088;       // B col = output n
  const bool fvalid = (lrow < 8);
  v8f acc = {};
  for (int kb = 0; kb < 784; ++kb) {
    v8h a0 = {};
    v8h a1 = {};
    if (fvalid) {
      a0 = *(const v8h*)(arow + kb * 32 + khalf * 8);
      a1 = *(const v8h*)(arow + kb * 32 + khalf * 8 + 16);
    }
    v16h a = __builtin_shufflevector(a0, a1, 0,1,2,3,4,5,6,7,8,9,10,11,12,13,14,15);
    v8h b0 = *(const v8h*)(brow + kb * 32 + khalf * 8);
    v8h b1 = *(const v8h*)(brow + kb * 32 + khalf * 8 + 16);
    v16h bm = __builtin_shufflevector(b0, b1, 0,1,2,3,4,5,6,7,8,9,10,11,12,13,14,15);
    acc = __builtin_amdgcn_wmma_f32_16x16x32_f16(false, a, false, bm,
                                                 (short)0, acc, false, false);
  }
  float bn = (n < 20) ? fc1b[n] : ((n < 220) ? fc4b[n - 20] : 0.f);
  for (int r = 0; r < 8; ++r) {
    int f = r + khalf * 8;
    if (f < 8 && n < 220) {
      float v = acc[r] + bn;
      if (n < 20) shape20[f * 20 + n] = v;                         // fc1: no relu
      else        pose200[f * 200 + (n - 20)] = fmaxf(v, 0.f);     // fc4: relu
    }
  }
}

// ------------------------------ small dense ops -----------------------------
__global__ void mean20(const float* __restrict__ s20, float* __restrict__ betas) {
  int c = threadIdx.x;
  if (c >= 20) return;
  float a = 0.f;
  for (int f = 0; f < 8; ++f) a += s20[f * 20 + c];
  betas[c] = a * 0.125f;
}

__global__ void matvec(const float* __restrict__ in, const float* __restrict__ W,
                       const float* __restrict__ b, float* __restrict__ out,
                       int B, int I, int O, int dorelu) {
  int i = blockIdx.x * blockDim.x + threadIdx.x;
  if (i >= B * O) return;
  int o = i % O, bb = i / O;
  float acc = b[o];
  const float* x = in + (size_t)bb * I;
  const float* w = W + (size_t)o * I;
  for (int k = 0; k < I; ++k) acc += x[k] * w[k];
  if (dorelu) acc = fmaxf(acc, 0.f);
  out[i] = acc;
}

__global__ void catk(const float* __restrict__ p200, const float* __restrict__ pf,
                     float* __restrict__ cat) {
  int i = blockIdx.x * blockDim.x + threadIdx.x;
  if (i >= 8 * 400) return;
  int c = i % 400, f = i / 400;
  cat[i] = (c < 200) ? p200[f * 200 + c] : pf[f * 200 + (c - 200)];
}

__global__ void fc2k(const float* __restrict__ betas, const float* __restrict__ W,
                     const float* __restrict__ b, float* __restrict__ out) {
  int i = blockIdx.x * blockDim.x + threadIdx.x;
  if (i >= 55168) return;
  float acc = b[i];
  const float* w = W + (size_t)i * 20;
  for (int k = 0; k < 20; ++k) acc += w[k] * betas[k];
  out[i] = acc;
}

// Rodrigues on pose_mean (first joint zeroed) -> pct[219] = [trans | 24 x 3x3]
__global__ void rodrigues(const float* __restrict__ pm, const float* __restrict__ trans,
                          float* __restrict__ pct) {
  int j = threadIdx.x;
  if (j < 3) pct[j] = trans[j];
  if (j >= 24) return;
  float rx = (j == 0) ? 0.f : pm[3 * j + 0];
  float ry = (j == 0) ? 0.f : pm[3 * j + 1];
  float rz = (j == 0) ? 0.f : pm[3 * j + 2];
  float th = sqrtf(rx * rx + ry * ry + rz * rz) + 1e-8f;
  float kx = rx / th, ky = ry / th, kz = rz / th;
  float s = sinf(th), oc = 1.f - cosf(th);
  float K[9]  = {0.f, -kz, ky,  kz, 0.f, -kx,  -ky, kx, 0.f};
  float K2[9];
  for (int r = 0; r < 3; ++r)
    for (int c = 0; c < 3; ++c) {
      float a = 0.f;
      for (int q = 0; q < 3; ++q) a += K[r * 3 + q] * K[q * 3 + c];
      K2[r * 3 + c] = a;
    }
  for (int e = 0; e < 9; ++e) {
    float I = (e == 0 || e == 4 || e == 8) ? 1.f : 0.f;
    pct[3 + j * 9 + e] = I + s * K[e] + oc * K2[e];
  }
}

// add = pose_results + pct; split into add_poses / add_trans regions of d_out.
__global__ void addout(const float* __restrict__ pres, const float* __restrict__ pct,
                       float* __restrict__ dout) {
  int i = blockIdx.x * blockDim.x + threadIdx.x;
  if (i >= 8 * 219) return;
  int c = i % 219, f = i / 219;
  float v = pres[i] + pct[c];
  if (c < 3) dout[22408 + f * 3 + c] = v;           // add_trans
  else       dout[20680 + f * 216 + (c - 3)] = v;   // add_poses
}

// ------------------------------ graph network -------------------------------
// out[n][o] = act( b[o] + sum_k sum_f (sum_d val[k,n,d]*x[idx[k,n,d]][f]) * W[k][f][o] )
__global__ void gclk(const float* __restrict__ x, const int* __restrict__ idx,
                     const float* __restrict__ val, const float* __restrict__ W,
                     const float* __restrict__ b, float* __restrict__ out,
                     int N, int F, int O, int mode) {   // mode 0: relu, 1: tanh/10
  __shared__ float sh[256];
  __shared__ int   sidx[64];
  __shared__ float sval[64];
  int n = blockIdx.x;
  int t = threadIdx.x;                                  // 64 threads
  sidx[t] = idx[((size_t)(t >> 4) * N + n) * 16 + (t & 15)];
  sval[t] = val[((size_t)(t >> 4) * N + n) * 16 + (t & 15)];
  __syncthreads();
  for (int k = 0; k < 4; ++k)
    for (int f = t; f < F; f += 64) {
      float a = 0.f;
      for (int d = 0; d < 16; ++d)
        a += sval[k * 16 + d] * x[(size_t)sidx[k * 16 + d] * F + f];
      sh[k * F + f] = a;
    }
  __syncthreads();
  if (t < O) {
    float a = b[t];
    for (int k = 0; k < 4; ++k)
      for (int f = 0; f < F; ++f)
        a += sh[k * F + f] * W[((size_t)k * F + f) * O + t];
    a = (mode == 0) ? fmaxf(a, 0.f) : (tanhf(a) * 0.1f);
    out[(size_t)n * O + t] = a;
  }
}

__global__ void spmmk(const float* __restrict__ x, const int* __restrict__ ui,
                      const float* __restrict__ uv, float* __restrict__ out,
                      int N, int F) {
  int i = blockIdx.x * blockDim.x + threadIdx.x;
  if (i >= N * F) return;
  int f = i % F; int n = i / F;
  float a = 0.f;
  for (int j = 0; j < 4; ++j)
    a += uv[n * 4 + j] * x[(size_t)ui[n * 4 + j] * F + f];
  out[i] = a;
}

// ---------------------------------------------------------------------------
extern "C" void kernel_launch(void* const* d_in, const int* in_sizes, int n_in,
                              void* d_out, int out_size, void* d_ws, size_t ws_size,
                              hipStream_t stream) {
  (void)in_sizes; (void)n_in; (void)out_size; (void)ws_size;
  const float* seg   = (const float*)d_in[0];
  const float* poses = (const float*)d_in[1];
  const float* pmean = (const float*)d_in[2];
  const float* trans = (const float*)d_in[3];
  const float* c1w = (const float*)d_in[4];  const float* c1b = (const float*)d_in[5];
  const float* c2w = (const float*)d_in[6];  const float* c2b = (const float*)d_in[7];
  const float* c3w = (const float*)d_in[8];  const float* c3b = (const float*)d_in[9];
  const float* c4w = (const float*)d_in[10]; const float* c4b = (const float*)d_in[11];
  const float* c5w = (const float*)d_in[12]; const float* c5b = (const float*)d_in[13];
  const float* fc1w = (const float*)d_in[14]; const float* fc1b = (const float*)d_in[15];
  const float* fc2w = (const float*)d_in[16]; const float* fc2b = (const float*)d_in[17];
  const float* fc3w = (const float*)d_in[18]; const float* fc3b = (const float*)d_in[19];
  const float* fc4w = (const float*)d_in[20]; const float* fc4b = (const float*)d_in[21];
  const float* fc5w = (const float*)d_in[22]; const float* fc5b = (const float*)d_in[23];
  const float* fc6w = (const float*)d_in[24]; const float* fc6b = (const float*)d_in[25];
  const float* fc7w = (const float*)d_in[26]; const float* fc7b = (const float*)d_in[27];
  const float* gW3 = (const float*)d_in[28]; const float* gb3 = (const float*)d_in[29];
  const float* gW2 = (const float*)d_in[30]; const float* gb2 = (const float*)d_in[31];
  const float* gW1 = (const float*)d_in[32]; const float* gb1 = (const float*)d_in[33];
  const float* gW0 = (const float*)d_in[34]; const float* gb0 = (const float*)d_in[35];
  const float* av3 = (const float*)d_in[36]; const int* ai3 = (const int*)d_in[37];
  const float* av2 = (const float*)d_in[38]; const int* ai2 = (const int*)d_in[39];
  const float* av1 = (const float*)d_in[40]; const int* ai1 = (const int*)d_in[41];
  const float* av0 = (const float*)d_in[42]; const int* ai0 = (const int*)d_in[43];
  const float* uv2 = (const float*)d_in[44]; const int* ui2 = (const int*)d_in[45];
  const float* uv1 = (const float*)d_in[46]; const int* ui1 = (const int*)d_in[47];
  const float* uv0 = (const float*)d_in[48]; const int* ui0 = (const int*)d_in[49];
  float* dout = (float*)d_out;  // [offsets 20670 | betas 10 | poses 1728 | trans 24]

  // ---- workspace bump allocator (~50 MB total) ----
  char* base = (char*)d_ws;
  size_t off = 0;
  auto alloc = [&](size_t bytes) -> void* {
    void* p = base + off;
    off = (off + bytes + 255) & ~(size_t)255;
    return p;
  };
  _Float16* actA  = (_Float16*)alloc((size_t)8 * 254 * 254 * 32 * 2);  // act ping (33.0 MB)
  _Float16* rawB  = (_Float16*)alloc((size_t)8 * 252 * 252 * 16 * 2);  // raw pong (16.3 MB)
  _Float16* wcat  = rawB;  // reused after conv stack (needs 11.2 MB <= 16.3 MB)
  _Float16* wp2   = (_Float16*)alloc((size_t)9 * 16 * 32 * 2);
  _Float16* wp3   = (_Float16*)alloc((size_t)9 * 32 * 32 * 2);
  _Float16* wp4   = (_Float16*)alloc((size_t)9 * 64 * 32 * 2);
  _Float16* wp5   = (_Float16*)alloc((size_t)9 * 128 * 64 * 2);
  _Float16* flatA = (_Float16*)alloc((size_t)8 * 25088 * 2);
  float* shape20  = (float*)alloc(8 * 20 * 4);
  float* pose200  = (float*)alloc(8 * 200 * 4);
  float* pfeat    = (float*)alloc(8 * 200 * 4);
  float* catb     = (float*)alloc(8 * 400 * 4);
  float* t6       = (float*)alloc(8 * 100 * 4);
  float* pres     = (float*)alloc(8 * 219 * 4);
  float* pct      = (float*)alloc(219 * 4);
  float* betas    = (float*)alloc(20 * 4);
  float* xg0      = (float*)alloc((size_t)55168 * 4);
  float* xg1      = (float*)alloc((size_t)862 * 32 * 4);
  float* xg2      = (float*)alloc((size_t)1723 * 32 * 4);
  float* xg3      = (float*)alloc((size_t)1723 * 16 * 4);
  float* xg4      = (float*)alloc((size_t)3445 * 16 * 4);
  float* xg5      = (float*)alloc((size_t)3445 * 16 * 4);
  float* xg6      = (float*)alloc((size_t)6890 * 16 * 4);

  // ---- repack conv weights to f16 B-matrix layout ----
  repack_conv_w<<<(9 * 16 * 32 + 255) / 256, 256, 0, stream>>>(c2w, wp2, 8, 32, 16);
  repack_conv_w<<<(9 * 32 * 32 + 255) / 256, 256, 0, stream>>>(c3w, wp3, 16, 32, 32);
  repack_conv_w<<<(9 * 64 * 32 + 255) / 256, 256, 0, stream>>>(c4w, wp4, 32, 32, 64);
  repack_conv_w<<<(9 * 128 * 64 + 255) / 256, 256, 0, stream>>>(c5w, wp5, 64, 64, 128);

  // ---- CNN base net (8 frames) ----
  conv1_pool<<<(8 * 254 * 254 + 255) / 256, 256, 0, stream>>>(seg, c1w, c1b, actA);

  conv_wmma<<<dim3(16, 252, 8),  32, 0, stream>>>(actA, wp2, c2b, rawB, 254, 32, 16, 252, 1);
  pool2<<<(int)(((size_t)8 * 126 * 126 * 32 + 255) / 256), 256, 0, stream>>>(rawB, actA, 252, 16, 126, 32);

  conv_wmma<<<dim3(8, 124, 16),  32, 0, stream>>>(actA, wp3, c3b, rawB, 126, 32, 32, 124, 1);
  pool2<<<(int)(((size_t)8 * 62 * 62 * 32 + 255) / 256), 256, 0, stream>>>(rawB, actA, 124, 32, 62, 32);

  conv_wmma<<<dim3(4, 60, 32),   32, 0, stream>>>(actA, wp4, c4b, rawB, 62, 32, 64, 60, 1);
  pool2<<<(int)(((size_t)8 * 30 * 30 * 64 + 255) / 256), 256, 0, stream>>>(rawB, actA, 60, 64, 30, 64);

  conv_wmma<<<dim3(2, 28, 64),   32, 0, stream>>>(actA, wp5, c5b, rawB, 30, 64, 128, 28, 0);
  pool2<<<(int)(((size_t)8 * 14 * 14 * 128 + 255) / 256), 256, 0, stream>>>(rawB, actA, 28, 128, 14, 128);

  flatten5<<<(8 * 25088 + 255) / 256, 256, 0, stream>>>(actA, flatA);

  // ---- fc1 + fc4 via WMMA GEMM (rawB reused as f16 weight buffer) ----
  repack_wcat<<<(int)(((size_t)224 * 25088 + 255) / 256), 256, 0, stream>>>(fc1w, fc4w, wcat);
  fcgemm<<<14, 32, 0, stream>>>(flatA, wcat, fc1b, fc4b, shape20, pose200);

  // ---- pose head ----
  matvec<<<(8 * 200 + 255) / 256, 256, 0, stream>>>(poses, fc5w, fc5b, pfeat, 8, 75, 200, 1);
  catk<<<(8 * 400 + 255) / 256, 256, 0, stream>>>(pose200, pfeat, catb);
  matvec<<<(8 * 100 + 255) / 256, 256, 0, stream>>>(catb, fc6w, fc6b, t6, 8, 400, 100, 0);
  matvec<<<(8 * 219 + 255) / 256, 256, 0, stream>>>(t6, fc7w, fc7b, pres, 8, 100, 219, 0);
  rodrigues<<<1, 32, 0, stream>>>(pmean, trans, pct);
  addout<<<(8 * 219 + 255) / 256, 256, 0, stream>>>(pres, pct, dout);

  // ---- shape head ----
  mean20<<<1, 32, 0, stream>>>(shape20, betas);
  matvec<<<1, 32, 0, stream>>>(betas, fc3w, fc3b, dout + 20670, 1, 20, 10, 0);  // betas_val
  fc2k<<<(55168 + 255) / 256, 256, 0, stream>>>(betas, fc2w, fc2b, xg0);

  // ---- graph conv / upsample chain -> offsets in d_out[0..20669] ----
  gclk<<<862, 64, 0, stream>>>(xg0, ai3, av3, gW3, gb3, xg1, 862, 64, 32, 0);
  spmmk<<<(1723 * 32 + 255) / 256, 256, 0, stream>>>(xg1, ui2, uv2, xg2, 1723, 32);
  gclk<<<1723, 64, 0, stream>>>(xg2, ai2, av2, gW2, gb2, xg3, 1723, 32, 16, 0);
  spmmk<<<(3445 * 16 + 255) / 256, 256, 0, stream>>>(xg3, ui1, uv1, xg4, 3445, 16);
  gclk<<<3445, 64, 0, stream>>>(xg4, ai1, av1, gW1, gb1, xg5, 3445, 16, 16, 0);
  spmmk<<<(6890 * 16 + 255) / 256, 256, 0, stream>>>(xg5, ui0, uv0, xg6, 6890, 16);
  gclk<<<6890, 64, 0, stream>>>(xg6, ai0, av0, gW0, gb0, dout, 6890, 16, 3, 1);
}
